// biLstmCrf_51625506898421
// MI455X (gfx1250) — matrix operand order repaired and checked by reading, not compiled, not run on hardware
//
#include <hip/hip_runtime.h>
#include <cstdint>
#include <cstddef>

// ---------------- problem constants ----------------
#define B_   32
#define S_   512
#define E_   256
#define H_   512
#define L_   48
#define G4H  2048            // 4*H
#define NT_HH 128            // 4H / 16  (N tiles of recurrent GEMM)
#define KT_C  24             // (H + E) / 32  (combined K tiles: 16 for h, 8 for x)
#define NT_EM 3              // L / 16
#define KT_EM 32             // 2H / 32

typedef __attribute__((ext_vector_type(16))) _Float16 v16h;
typedef __attribute__((ext_vector_type(8)))  _Float16 v8h;
typedef __attribute__((ext_vector_type(8)))  float    v8f;

// Global-address-space pointer types: keep the opaque weight pointer
// provably global so loads lower to global_load_b128 (LOADcnt only),
// not flat_load_b128 (LOADcnt + DScnt coupling with the LDS stream).
#define GAS __attribute__((address_space(1)))
typedef const _Float16 GAS* ghp;
typedef const v16h     GAS* gv16hp;

// =====================================================================
// Pack [W_hh | W_ih] (f32, row-major (4H,H)+(4H,E)) into f16 WMMA
// B-fragment order: dst[((nt*KT_C + kk)*512) + lane*16 + j]
//   = W^T[32*kk + (j>>3)*16 + (lane>=16?8:0) + (j&7)][16*nt + (lane&15)]
// =====================================================================
__global__ void pack_wcomb_kernel(const float* __restrict__ whh_f,
                                  const float* __restrict__ wih_f,
                                  const float* __restrict__ whh_b,
                                  const float* __restrict__ wih_b,
                                  _Float16* __restrict__ dst) {
  const int PER_DIR = NT_HH * KT_C * 512;                 // 1,572,864
  int gid = blockIdx.x * blockDim.x + threadIdx.x;
  if (gid >= 2 * PER_DIR) return;
  int dir  = gid / PER_DIR;
  int rem  = gid - dir * PER_DIR;
  int slab = rem >> 9;
  int kk   = slab % KT_C;
  int nt   = slab / KT_C;
  int lj   = rem & 511;
  int lane = lj >> 4;
  int j    = lj & 15;
  int n = nt * 16 + (lane & 15);
  int k = kk * 32 + ((j >> 3) * 16) + ((lane >= 16) ? 8 : 0) + (j & 7);
  const float* whh = dir ? whh_b : whh_f;
  const float* wih = dir ? wih_b : wih_f;
  float v = (k < H_) ? whh[n * H_ + k] : wih[n * E_ + (k - H_)];
  dst[gid] = (_Float16)v;
}

// Pack W_em (L, 2H) into the same B-fragment order.
__global__ void pack_wem_kernel(const float* __restrict__ wem,
                                _Float16* __restrict__ dst) {
  const int TOT = NT_EM * KT_EM * 512;                    // 49,152
  int gid = blockIdx.x * blockDim.x + threadIdx.x;
  if (gid >= TOT) return;
  int slab = gid >> 9;
  int kk   = slab % KT_EM;
  int nt   = slab / KT_EM;
  int lj   = gid & 511;
  int lane = lj >> 4;
  int j    = lj & 15;
  int n = nt * 16 + (lane & 15);
  int k = kk * 32 + ((j >> 3) * 16) + ((lane >= 16) ? 8 : 0) + (j & 7);
  dst[gid] = (_Float16)wem[n * (2 * H_) + k];
}

// Gather embeddings to f16, row-major [B][S][E].
__global__ void embed_gather_kernel(const float* __restrict__ tab,
                                    const int* __restrict__ tokens,
                                    _Float16* __restrict__ xe) {
  int gid = blockIdx.x * blockDim.x + threadIdx.x;        // 4 elems / thread
  if (gid >= B_ * S_ * E_ / 4) return;
  int idx4 = gid * 4;
  int bs = idx4 >> 8;                                     // E == 256
  int e0 = idx4 & 255;
  int tok = tokens[bs];
  const float4 v = *(const float4*)(tab + (size_t)tok * E_ + e0);
  _Float16* d = xe + idx4;
  d[0] = (_Float16)v.x; d[1] = (_Float16)v.y;
  d[2] = (_Float16)v.z; d[3] = (_Float16)v.w;
}

// =====================================================================
// Fused LSTM scan, one workgroup per (direction, batch chunk of 16).
// gates(16x2048) = [h(16x512) | x_t(16x256)] @ Wcomb^T + b, per step.
// 32 waves; wave w owns h columns [16w,16w+16) and computes gate tiles
// {w, 32+w, 64+w, 96+w}, i.e. i/f/g/o for exactly its columns.
// Weights stream from L2 every step; an empty asm barrier on an
// addrspace(1) pointer makes the base opaque per iteration (no LICM/CSE
// -> no 768-VGPR spill) while keeping loads on the GLOBAL path.
// Cell state c lives entirely in registers (static lane ownership).
// =====================================================================
__device__ __forceinline__ float sigm_(float x) { return 1.0f / (1.0f + expf(-x)); }

__global__ void __launch_bounds__(1024)
lstm_scan_kernel(const _Float16* __restrict__ xe,      // [B][S][E] f16
                 const _Float16* __restrict__ wcomb,   // [2][NT_HH*KT_C*512] f16
                 const float* __restrict__ bias_f,
                 const float* __restrict__ bias_b,
                 _Float16* __restrict__ hs16) {        // [2][B][S][H] f16
  const int HST = 520;   // padded f16 row stride (1040B, not a 256B multiple)
  __shared__ _Float16 h16[16 * HST];

  const int dir   = blockIdx.x & 1;
  const int bbase = (blockIdx.x >> 1) * 16;
  const int tid   = threadIdx.x;
  const int lane  = tid & 31;
  const int w     = tid >> 5;                 // wave id 0..31
  const int ln15  = lane & 15;
  const int hi    = (lane >= 16) ? 1 : 0;
  const int base  = hi ? 8 : 0;
  const int col   = w * 16 + ln15;            // this wave's h column for this lane

  for (int i = tid; i < 16 * HST; i += 1024) h16[i] = (_Float16)0.f;

  const float* bvec = dir ? bias_b : bias_f;
  const float bi0 = bvec[0 * H_ + col], bi1 = bvec[1 * H_ + col];
  const float bi2 = bvec[2 * H_ + col], bi3 = bvec[3 * H_ + col];

  ghp wb = (ghp)(wcomb + (size_t)dir * (NT_HH * KT_C * 512));
  const size_t boff0 = ((size_t)(0 * 32 + w) * KT_C) * 512 + lane * 16;
  const size_t boff1 = ((size_t)(1 * 32 + w) * KT_C) * 512 + lane * 16;
  const size_t boff2 = ((size_t)(2 * 32 + w) * KT_C) * 512 + lane * 16;
  const size_t boff3 = ((size_t)(3 * 32 + w) * KT_C) * 512 + lane * 16;

  // cell state in registers: lane owns rows (r + 8*hi) of its column
  float creg[8];
#pragma unroll
  for (int r = 0; r < 8; ++r) creg[r] = 0.f;

  __syncthreads();

  for (int t = 0; t < S_; ++t) {
    const int tt = dir ? (S_ - 1 - t) : t;

    // Opaque copy of the weight base each iteration: blocks LICM/CSE so
    // B-fragments are (re)loaded from L2 inside the loop, not spilled.
    ghp wbt = wb;
    asm volatile("" : "+s"(wbt));

    v8f a0, a1, a2, a3;
#pragma unroll
    for (int r = 0; r < 8; ++r) { a0[r] = bi0; a1[r] = bi1; a2[r] = bi2; a3[r] = bi3; }

    const _Float16* xrow = xe + ((size_t)(bbase + ln15) * S_ + tt) * E_;
    if (t + 1 < S_) {
      const int tn = dir ? (S_ - 2 - t) : (t + 1);
      __builtin_prefetch(xe + ((size_t)(bbase + ln15) * S_ + tn) * E_, 0, 1);
    }

    // --- recurrent part: K = 0..511, A from h in LDS ---
    for (int kk = 0; kk < 16; ++kk) {
      const _Float16* hp = &h16[ln15 * HST + kk * 32 + base];
      v8h lo  = *(const v8h*)(hp);
      v8h hi8 = *(const v8h*)(hp + 16);
      v16h a;
#pragma unroll
      for (int j = 0; j < 8; ++j) { a[j] = lo[j]; a[8 + j] = hi8[j]; }
      ghp bp = wbt + (size_t)kk * 512;
      v16h b0 = *(gv16hp)(bp + boff0);
      v16h b1 = *(gv16hp)(bp + boff1);
      v16h b2 = *(gv16hp)(bp + boff2);
      v16h b3 = *(gv16hp)(bp + boff3);
      a0 = __builtin_amdgcn_wmma_f32_16x16x32_f16(false, a, false, b0, (short)0, a0, false, false);
      a1 = __builtin_amdgcn_wmma_f32_16x16x32_f16(false, a, false, b1, (short)0, a1, false, false);
      a2 = __builtin_amdgcn_wmma_f32_16x16x32_f16(false, a, false, b2, (short)0, a2, false, false);
      a3 = __builtin_amdgcn_wmma_f32_16x16x32_f16(false, a, false, b3, (short)0, a3, false, false);
    }
    // --- input-projection part: K = 512..767, A from x_t in global (L2-hot) ---
    for (int kk = 16; kk < KT_C; ++kk) {
      const _Float16* xp = xrow + (kk - 16) * 32 + base;
      v8h lo  = *(const v8h*)(xp);
      v8h hi8 = *(const v8h*)(xp + 16);
      v16h a;
#pragma unroll
      for (int j = 0; j < 8; ++j) { a[j] = lo[j]; a[8 + j] = hi8[j]; }
      ghp bp = wbt + (size_t)kk * 512;
      v16h b0 = *(gv16hp)(bp + boff0);
      v16h b1 = *(gv16hp)(bp + boff1);
      v16h b2 = *(gv16hp)(bp + boff2);
      v16h b3 = *(gv16hp)(bp + boff3);
      a0 = __builtin_amdgcn_wmma_f32_16x16x32_f16(false, a, false, b0, (short)0, a0, false, false);
      a1 = __builtin_amdgcn_wmma_f32_16x16x32_f16(false, a, false, b1, (short)0, a1, false, false);
      a2 = __builtin_amdgcn_wmma_f32_16x16x32_f16(false, a, false, b2, (short)0, a2, false, false);
      a3 = __builtin_amdgcn_wmma_f32_16x16x32_f16(false, a, false, b3, (short)0, a3, false, false);
    }

    __syncthreads();   // all waves done reading h16 of step t

    // elementwise LSTM cell update: D-frag lane holds (M = r + 8*hi, N = ln15)
#pragma unroll
    for (int r = 0; r < 8; ++r) {
      const int mm = r + hi * 8;
      float iv = sigm_(a0[r]);
      float fv = sigm_(a1[r]);
      float gv = tanhf(a2[r]);
      float ov = sigm_(a3[r]);
      float c  = fv * creg[r] + iv * gv;
      float h  = ov * tanhf(c);
      creg[r] = c;
      h16[mm * HST + col] = (_Float16)h;
      hs16[(((size_t)(dir * B_ + bbase + mm)) * S_ + tt) * H_ + col] = (_Float16)h;
    }
    __syncthreads();   // new h16 visible before step t+1
  }
}

// =====================================================================
// Emission: e = exp(concat(hf,hb) @ W_em^T + b_em), WMMA, N = 48 = 3 tiles.
// One wave per M-tile of 16 rows of (B*S).
// =====================================================================
__global__ void __launch_bounds__(256)
emission_kernel(const _Float16* __restrict__ hs16,    // [2][B][S][H]
                const _Float16* __restrict__ wemp,    // packed [3*32*512]
                const float* __restrict__ bem,
                float* __restrict__ ebuf) {           // [B*S][L]
  const int tid  = threadIdx.x;
  const int lane = tid & 31;
  const int wid  = tid >> 5;
  const int mt   = blockIdx.x * 8 + wid;              // 0..1023
  const int ln15 = lane & 15;
  const int hi   = (lane >= 16) ? 1 : 0;
  const int base = hi ? 8 : 0;
  const int rr   = mt * 16 + ln15;
  const int b    = rr >> 9;
  const int s    = rr & 511;

  v8f acc0, acc1, acc2;
  float b0 = bem[ln15], b1 = bem[16 + ln15], b2 = bem[32 + ln15];
#pragma unroll
  for (int r = 0; r < 8; ++r) { acc0[r] = b0; acc1[r] = b1; acc2[r] = b2; }

  for (int kk = 0; kk < KT_EM; ++kk) {
    const int dir  = kk >> 4;                          // k<512 -> hf, else hb
    const int koff = (kk & 15) * 32 + base;
    const _Float16* ap = hs16 + (((size_t)(dir * B_ + b)) * S_ + s) * H_ + koff;
    v8h lo  = *(const v8h*)(ap);
    v8h hi8 = *(const v8h*)(ap + 16);
    v16h a;
#pragma unroll
    for (int j = 0; j < 8; ++j) { a[j] = lo[j]; a[8 + j] = hi8[j]; }
    const size_t ko = (size_t)kk * 512 + lane * 16;
    v16h bf0 = *(const v16h*)(wemp + (size_t)(0 * KT_EM) * 512 + ko);
    v16h bf1 = *(const v16h*)(wemp + (size_t)(1 * KT_EM) * 512 + ko);
    v16h bf2 = *(const v16h*)(wemp + (size_t)(2 * KT_EM) * 512 + ko);
    acc0 = __builtin_amdgcn_wmma_f32_16x16x32_f16(false, a, false, bf0, (short)0, acc0, false, false);
    acc1 = __builtin_amdgcn_wmma_f32_16x16x32_f16(false, a, false, bf1, (short)0, acc1, false, false);
    acc2 = __builtin_amdgcn_wmma_f32_16x16x32_f16(false, a, false, bf2, (short)0, acc2, false, false);
  }
#pragma unroll
  for (int r = 0; r < 8; ++r) {
    const size_t row = (size_t)mt * 16 + r + hi * 8;
    ebuf[row * L_ +  0 + ln15] = expf(acc0[r]);
    ebuf[row * L_ + 16 + ln15] = expf(acc1[r]);
    ebuf[row * L_ + 32 + ln15] = expf(acc2[r]);
  }
}

// =====================================================================
// CRF forward recursion + gold score. One 64-thread block per batch.
// new_alpha[i] = LSE_j(alpha[i] + Te[i][j] + e_t[j]) -> alpha[i] stays
// in a register of thread i (no cross-thread alpha reads).
// =====================================================================
__global__ void __launch_bounds__(64)
crf_score_kernel(const float* __restrict__ ebuf,
                 const float* __restrict__ trans,
                 const int* __restrict__ labels,
                 const int* __restrict__ length,
                 float* __restrict__ out) {
  const int b = blockIdx.x;
  const int tid = threadIdx.x;
  __shared__ float Te[L_ * L_];
  __shared__ float et[L_];
  __shared__ float red[64];
  __shared__ float ash[L_];

  for (int i = tid; i < L_ * L_; i += 64) Te[i] = expf(trans[i]);
  __syncthreads();

  const int len = length[b];
  const int* lab = labels + b * S_;
  const float* eb = ebuf + (size_t)b * S_ * L_;

  float racc = 0.f;
  for (int t = tid; t < S_; t += 64) {
    if (t < len) {
      racc += eb[t * L_ + lab[t]];
      if (t >= 1) racc += Te[lab[t - 1] * L_ + lab[t]];
    }
  }
  red[tid] = racc;

  float alpha = (tid < L_) ? eb[tid] : 0.f;
  for (int t = 1; t < S_; ++t) {
    __syncthreads();
    if (tid < L_) et[tid] = eb[t * L_ + tid];
    __syncthreads();
    if (tid < L_ && t < len) {
      const float* Ti = &Te[tid * L_];
      float vmax = -3.402823466e38f;
      for (int j = 0; j < L_; ++j) vmax = fmaxf(vmax, alpha + Ti[j] + et[j]);
      float sacc = 0.f;
      for (int j = 0; j < L_; ++j) sacc += expf(alpha + Ti[j] + et[j] - vmax);
      alpha = vmax + logf(sacc);
    }
  }
  __syncthreads();
  if (tid < L_) ash[tid] = alpha;
  __syncthreads();
  if (tid == 0) {
    float rs = 0.f;
    for (int i = 0; i < 64; ++i) rs += red[i];
    float am = -3.402823466e38f;
    for (int i = 0; i < L_; ++i) am = fmaxf(am, ash[i]);
    float ss = 0.f;
    for (int i = 0; i < L_; ++i) ss += expf(ash[i] - am);
    out[b] = am + logf(ss) - rs;
  }
}

// =====================================================================
extern "C" void kernel_launch(void* const* d_in, const int* in_sizes, int n_in,
                              void* d_out, int out_size, void* d_ws, size_t ws_size,
                              hipStream_t stream) {
  (void)in_sizes; (void)n_in; (void)out_size; (void)ws_size;
  const int*   tokens = (const int*)d_in[0];
  const int*   length = (const int*)d_in[1];
  const int*   labels = (const int*)d_in[2];
  const float* embed  = (const float*)d_in[3];
  const float* wih_f  = (const float*)d_in[4];
  const float* whh_f  = (const float*)d_in[5];
  const float* b_f    = (const float*)d_in[6];
  const float* wih_b  = (const float*)d_in[7];
  const float* whh_b  = (const float*)d_in[8];
  const float* b_b    = (const float*)d_in[9];
  const float* w_em   = (const float*)d_in[10];
  const float* b_em   = (const float*)d_in[11];
  const float* trans  = (const float*)d_in[12];
  float* out = (float*)d_out;

  char* ws = (char*)d_ws;
  size_t off = 0;
  _Float16* wcomb = (_Float16*)(ws + off); off += (size_t)2 * NT_HH * KT_C * 512 * 2; // 6.0 MB
  _Float16* wemp  = (_Float16*)(ws + off); off += (size_t)NT_EM * KT_EM * 512 * 2;    // 96 KB
  off = (off + 255) & ~(size_t)255;
  _Float16* xe16  = (_Float16*)(ws + off); off += (size_t)B_ * S_ * E_ * 2;           // 8.4 MB
  _Float16* hs16  = (_Float16*)(ws + off); off += (size_t)2 * B_ * S_ * H_ * 2;       // 33.6 MB
  float*    ebuf  = (float*)(ws + off);    off += (size_t)B_ * S_ * L_ * 4;           // 3.1 MB

  pack_wcomb_kernel<<<(2 * NT_HH * KT_C * 512) / 256, 256, 0, stream>>>(
      whh_f, wih_f, whh_b, wih_b, wcomb);
  pack_wem_kernel<<<(NT_EM * KT_EM * 512) / 256, 256, 0, stream>>>(w_em, wemp);
  embed_gather_kernel<<<(B_ * S_ * E_ / 4) / 256, 256, 0, stream>>>(embed, tokens, xe16);
  lstm_scan_kernel<<<4, 1024, 0, stream>>>(xe16, wcomb, b_f, b_b, hs16);
  emission_kernel<<<128, 256, 0, stream>>>(hs16, wemp, b_em, ebuf);
  crf_score_kernel<<<B_, 64, 0, stream>>>(ebuf, trans, labels, length, out);
}
